// StandardTransformerBlock_64871186039066
// MI455X (gfx1250) — compile-verified, matrix operand order checked by reading, not run on hardware
//
#include <hip/hip_runtime.h>
#include <hip/hip_bf16.h>

// ---------------------------------------------------------------------------
// StandardTransformerBlock for MI455X (gfx1250), wave32 + WMMA bf16 path.
// B=2, T=2048, D=1024, H=16, HD=64.  All GEMMs and both attention einsums
// run on v_wmma_f32_16x16x32_bf16 (f32 accumulate).
// Round-4: clean pair-packed V-transpose staging in attention (8 ds_store_b32
// per thread instead of 16 ds_store_b16); GEMM keeps register-prefetch double
// buffering and pair-packed B-tile transpose.
// ---------------------------------------------------------------------------

typedef __bf16 bf16_t;
typedef __attribute__((ext_vector_type(2)))  __bf16 v2bf;
typedef __attribute__((ext_vector_type(8)))  __bf16 v8bf;
typedef __attribute__((ext_vector_type(16))) __bf16 v16bf;
typedef __attribute__((ext_vector_type(8)))  float  v8f;

#define D_MODEL 1024
#define N_HEAD  16
#define HEAD_D  64
#define SEQ_T   2048
#define BATCH   2
#define M_TOK   (BATCH * SEQ_T)   // 4096

__device__ __forceinline__ bf16_t f2bf(float f) {
    union { float f; unsigned u; } v; v.f = f;
    unsigned r = (v.u + 0x7FFFu + ((v.u >> 16) & 1u)) >> 16;   // RNE
    union { unsigned short s; bf16_t b; } o; o.s = (unsigned short)r;
    return o.b;
}

__device__ __forceinline__ v16bf cat8(v8bf lo, v8bf hi) {
    return __builtin_shufflevector(lo, hi, 0,1,2,3,4,5,6,7,8,9,10,11,12,13,14,15);
}

// ---------------------------------------------------------------------------
// fp32 -> bf16 bulk convert (weights)
// ---------------------------------------------------------------------------
__global__ __launch_bounds__(256)
void cvt_f32_bf16(const float* __restrict__ in, bf16_t* __restrict__ out, long n) {
    long i = (long)blockIdx.x * 256 + threadIdx.x;
    if (i < n) out[i] = f2bf(in[i]);
}

// ---------------------------------------------------------------------------
// RMSNorm: one 256-thread block per row of 1024; bf16 output.
// ---------------------------------------------------------------------------
__global__ __launch_bounds__(256)
void rmsnorm_kernel(const float* __restrict__ x, const float* __restrict__ scale,
                    bf16_t* __restrict__ out) {
    __shared__ float red[8];
    const long row = blockIdx.x;
    const float* xr = x + row * D_MODEL;
    float v[4], ss = 0.f;
#pragma unroll
    for (int i = 0; i < 4; i++) {
        v[i] = xr[threadIdx.x + i * 256];
        ss += v[i] * v[i];
    }
#pragma unroll
    for (int off = 16; off > 0; off >>= 1) ss += __shfl_xor(ss, off, 32);
    const int wave = threadIdx.x >> 5, lane = threadIdx.x & 31;
    if (lane == 0) red[wave] = ss;
    __syncthreads();
    float tot = 0.f;
#pragma unroll
    for (int i = 0; i < 8; i++) tot += red[i];
    const float rinv = rsqrtf(tot * (1.0f / D_MODEL) + 1e-8f);
#pragma unroll
    for (int i = 0; i < 4; i++) {
        const int c = threadIdx.x + i * 256;
        out[row * D_MODEL + c] = f2bf(v[i] * rinv * scale[c]);
    }
}

// ---------------------------------------------------------------------------
// Tiled bf16 WMMA GEMM: C[M,N] = A[M,K] x B[K,N], f32 accumulate.
// Block tile 128x128, K-tile 64, register-prefetch double buffering.
// 8 waves as 4x2; each wave owns a 32x64 patch (2x4 WMMA tiles, 16 WMMA/stage).
// EPI: 0 = store bf16, 1 = add fp32 residual -> fp32, 2 = exact GELU -> bf16
// ---------------------------------------------------------------------------
template<int EPI>
__global__ __launch_bounds__(256)
void gemm_bf16_wmma(const bf16_t* __restrict__ A, const bf16_t* __restrict__ B,
                    int M, int N, int K,
                    const float* __restrict__ resid,
                    float* __restrict__ outF, bf16_t* __restrict__ outB) {
    __shared__ __align__(16) bf16_t As[128][64];
    __shared__ __align__(16) bf16_t Bt[128][64];   // [n][k] (transposed tile)

    const int tid  = threadIdx.x;
    const int wave = tid >> 5;
    const int lane = tid & 31;
    const int lh   = lane >> 4;     // half-wave (selects K sub-chunk)
    const int ll   = lane & 15;
    const int m0   = blockIdx.y * 128;
    const int n0   = blockIdx.x * 128;
    const int wm   = (wave >> 1) * 32;   // 0,32,64,96
    const int wn   = (wave & 1) * 64;    // 0,64

    // staging coordinates
    const int ar  = tid >> 1;           // A row 0..127
    const int ac0 = (tid & 1) * 32;     // A col base (4 x 8 chunks)
    const int bkp = (tid >> 4) * 2;     // B k-pair base 0..30
    const int bc  = (tid & 15) * 8;     // B n col 0..120

    v8bf aR[4];        // A prefetch regs
    v8bf bR[2][2];     // B prefetch regs [kk-group][row-in-pair]

    auto loadTiles = [&](int k0) {
#pragma unroll
        for (int i = 0; i < 4; i++)
            aR[i] = *(const v8bf*)&A[(long)(m0 + ar) * K + k0 + ac0 + i * 8];
#pragma unroll
        for (int g = 0; g < 2; g++)
#pragma unroll
            for (int p = 0; p < 2; p++)
                bR[g][p] = *(const v8bf*)&B[(long)(k0 + bkp + p + g * 32) * N + n0 + bc];
    };
    auto storeTiles = [&]() {
#pragma unroll
        for (int i = 0; i < 4; i++)
            *(v8bf*)&As[ar][ac0 + i * 8] = aR[i];
#pragma unroll
        for (int g = 0; g < 2; g++)
#pragma unroll
            for (int j = 0; j < 8; j++) {
                v2bf pk;
                pk[0] = bR[g][0][j];
                pk[1] = bR[g][1][j];
                *(v2bf*)&Bt[bc + j][bkp + g * 32] = pk;   // ds_store_b32
            }
    };

    v8f acc[2][4];
#pragma unroll
    for (int i = 0; i < 2; i++)
#pragma unroll
        for (int j = 0; j < 4; j++)
#pragma unroll
            for (int e = 0; e < 8; e++) acc[i][j][e] = 0.f;

    loadTiles(0);
    storeTiles();
    __syncthreads();

    for (int k0 = 0;;) {
        const int  knext = k0 + 64;
        const bool more  = knext < K;
        if (more) loadTiles(knext);     // global loads overlap the WMMA stage

#pragma unroll
        for (int kk = 0; kk < 64; kk += 32) {
            v16bf af[2], bq[4];
#pragma unroll
            for (int i = 0; i < 2; i++) {
                const int row = wm + i * 16 + ll;
                v8bf lo = *(const v8bf*)&As[row][kk + lh * 8];
                v8bf hi = *(const v8bf*)&As[row][kk + 16 + lh * 8];
                af[i] = cat8(lo, hi);
            }
#pragma unroll
            for (int j = 0; j < 4; j++) {
                const int col = wn + j * 16 + ll;
                v8bf lo = *(const v8bf*)&Bt[col][kk + lh * 8];
                v8bf hi = *(const v8bf*)&Bt[col][kk + 16 + lh * 8];
                bq[j] = cat8(lo, hi);
            }
#pragma unroll
            for (int i = 0; i < 2; i++)
#pragma unroll
                for (int j = 0; j < 4; j++)
                    acc[i][j] = __builtin_amdgcn_wmma_f32_16x16x32_bf16(
                        false, af[i], false, bq[j], (short)0, acc[i][j], false, false);
        }
        __syncthreads();                // all waves done reading LDS
        if (!more) break;
        storeTiles();
        __syncthreads();
        k0 = knext;
    }

    // Epilogue
#pragma unroll
    for (int i = 0; i < 2; i++)
#pragma unroll
        for (int j = 0; j < 4; j++)
#pragma unroll
            for (int r = 0; r < 8; r++) {
                const int  row = m0 + wm + i * 16 + r + lh * 8;
                const int  col = n0 + wn + j * 16 + ll;
                const long idx = (long)row * N + col;
                float v = acc[i][j][r];
                if (EPI == 0) {
                    outB[idx] = f2bf(v);
                } else if (EPI == 1) {
                    outF[idx] = resid[idx] + v;
                } else {                       // exact GELU
                    v = 0.5f * v * (1.0f + erff(v * 0.70710678f));
                    outB[idx] = f2bf(v);
                }
            }
}

// ---------------------------------------------------------------------------
// Flash-style causal attention. Grid: (T/128, B*H). Block: 256 threads.
// Each wave owns 16 query rows; softmax stats reduce inside 16-lane halves.
// qkv layout: [B*T, 3*D] with inner [sel, H, HD].
// ---------------------------------------------------------------------------
__global__ __launch_bounds__(256)
void attention_wmma(const bf16_t* __restrict__ qkv, bf16_t* __restrict__ attn_out) {
    __shared__ __align__(16) bf16_t Qs[128][64];
    __shared__ __align__(16) bf16_t Ks[64][64];       // [key][d]  (== Bt for QK^T)
    __shared__ __align__(16) bf16_t Vt[64][64];       // [d][key]  (== Bt for PV)
    __shared__ __align__(16) bf16_t Ps[8][16][64];    // per-wave P staging

    const int tid  = threadIdx.x;
    const int wave = tid >> 5;
    const int lane = tid & 31;
    const int lh   = lane >> 4;
    const int ll   = lane & 15;
    const int bh   = blockIdx.y;
    const int b    = bh >> 4;            // / N_HEAD
    const int h    = bh & 15;
    const int q0   = blockIdx.x * 128;
    const long rs  = 3 * D_MODEL;        // qkv row stride (elements)

    const bf16_t* qb = qkv + (long)(b * SEQ_T) * rs + h * HEAD_D;
    const bf16_t* kb = qb + D_MODEL;
    const bf16_t* vb = qb + 2 * D_MODEL;

    // staging coords for K/V: each thread owns key-row pair (2*rr, 2*rr+1),
    // d-chunk cc.  256 threads == 32 row-pairs x 8 chunks == full 64x64 tile.
    const int rr = tid >> 3;             // 0..31  -> key rows 2rr, 2rr+1
    const int cc = (tid & 7) * 8;        // 0..56

    // Load Q tile 128x64
#pragma unroll
    for (int it = 0; it < 4; it++) {
        const int idx = tid + it * 256;
        const int r = idx >> 3, c = (idx & 7) * 8;
        *(v8bf*)&Qs[r][c] = *(const v8bf*)&qb[(long)(q0 + r) * rs + c];
    }

    float mi[8], li[8];
#pragma unroll
    for (int r = 0; r < 8; r++) { mi[r] = -3.0e38f; li[r] = 0.f; }
    v8f acc[4];
#pragma unroll
    for (int j = 0; j < 4; j++)
#pragma unroll
        for (int e = 0; e < 8; e++) acc[j][e] = 0.f;

    const int wq  = wave * 16;
    const int nkb = (q0 + 128) / 64;     // causal: key blocks covering <= q0+127
    __syncthreads();

    for (int kbi = 0; kbi < nkb; kbi++) {
        const int kb0 = kbi * 64;
        // Stage K (natural b128) and V (transposed, pair-packed b32 stores)
        {
            const int r0 = rr * 2;
            *(v8bf*)&Ks[r0][cc]     = *(const v8bf*)&kb[(long)(kb0 + r0) * rs + cc];
            *(v8bf*)&Ks[r0 + 1][cc] = *(const v8bf*)&kb[(long)(kb0 + r0 + 1) * rs + cc];
            v8bf va = *(const v8bf*)&vb[(long)(kb0 + r0) * rs + cc];
            v8bf vbv = *(const v8bf*)&vb[(long)(kb0 + r0 + 1) * rs + cc];
#pragma unroll
            for (int j = 0; j < 8; j++) {
                v2bf pk;
                pk[0] = va[j];
                pk[1] = vbv[j];
                *(v2bf*)&Vt[cc + j][r0] = pk;             // ds_store_b32
            }
        }
        __syncthreads();

        // S = Q K^T : 1x4 tiles per wave (rows wq..wq+15, 64 keys)
        v8f s[4];
#pragma unroll
        for (int nt = 0; nt < 4; nt++)
#pragma unroll
            for (int e = 0; e < 8; e++) s[nt][e] = 0.f;
#pragma unroll
        for (int kk = 0; kk < 64; kk += 32) {
            v8bf qlo = *(const v8bf*)&Qs[wq + ll][kk + lh * 8];
            v8bf qhi = *(const v8bf*)&Qs[wq + ll][kk + 16 + lh * 8];
            v16bf af = cat8(qlo, qhi);
#pragma unroll
            for (int nt = 0; nt < 4; nt++) {
                v8bf klo = *(const v8bf*)&Ks[nt * 16 + ll][kk + lh * 8];
                v8bf khi = *(const v8bf*)&Ks[nt * 16 + ll][kk + 16 + lh * 8];
                s[nt] = __builtin_amdgcn_wmma_f32_16x16x32_bf16(
                    false, af, false, cat8(klo, khi), (short)0, s[nt], false, false);
            }
        }

        // Online softmax (per row; rows r+8*lh live in lane-halves)
#pragma unroll
        for (int r = 0; r < 8; r++) {
            const int grow = q0 + wq + r + lh * 8;
            float pm = -3.0e38f;
#pragma unroll
            for (int nt = 0; nt < 4; nt++) {
                const int gcol = kb0 + nt * 16 + ll;
                float v = s[nt][r] * 0.125f;             // 1/sqrt(HD)
                v = (gcol <= grow) ? v : -3.0e38f;       // causal mask
                s[nt][r] = v;
                pm = fmaxf(pm, v);
            }
#pragma unroll
            for (int off = 1; off < 16; off <<= 1)
                pm = fmaxf(pm, __shfl_xor(pm, off, 32));
            const float mnew  = fmaxf(mi[r], pm);
            const float alpha = __expf(mi[r] - mnew);
            float ps = 0.f;
#pragma unroll
            for (int nt = 0; nt < 4; nt++) {
                const float p = __expf(s[nt][r] - mnew);
                s[nt][r] = p;
                ps += p;
            }
#pragma unroll
            for (int off = 1; off < 16; off <<= 1)
                ps += __shfl_xor(ps, off, 32);
            li[r] = li[r] * alpha + ps;
            mi[r] = mnew;
#pragma unroll
            for (int j = 0; j < 4; j++) acc[j][r] *= alpha;
        }

        // Reshape P (C layout) -> A layout via per-wave LDS slab
#pragma unroll
        for (int nt = 0; nt < 4; nt++)
#pragma unroll
            for (int r = 0; r < 8; r++)
                Ps[wave][r + lh * 8][nt * 16 + ll] = f2bf(s[nt][r]);

        // O += P V
#pragma unroll
        for (int kk = 0; kk < 64; kk += 32) {
            v8bf plo = *(const v8bf*)&Ps[wave][ll][kk + lh * 8];
            v8bf phi = *(const v8bf*)&Ps[wave][ll][kk + 16 + lh * 8];
            v16bf pf = cat8(plo, phi);
#pragma unroll
            for (int dt = 0; dt < 4; dt++) {
                v8bf vlo = *(const v8bf*)&Vt[dt * 16 + ll][kk + lh * 8];
                v8bf vhi = *(const v8bf*)&Vt[dt * 16 + ll][kk + 16 + lh * 8];
                acc[dt] = __builtin_amdgcn_wmma_f32_16x16x32_bf16(
                    false, pf, false, cat8(vlo, vhi), (short)0, acc[dt], false, false);
            }
        }
        __syncthreads();
    }

    // Normalize and store: attn_out[b*T+row][h*64+d], bf16
#pragma unroll
    for (int r = 0; r < 8; r++) {
        const float inv = 1.0f / li[r];
        const int grow  = q0 + wq + r + lh * 8;
        const long orow = (long)(b * SEQ_T + grow) * D_MODEL + h * HEAD_D;
#pragma unroll
        for (int dt = 0; dt < 4; dt++)
            attn_out[orow + dt * 16 + ll] = f2bf(acc[dt][r] * inv);
    }
}

// ---------------------------------------------------------------------------
// Host-side orchestration
// ---------------------------------------------------------------------------
extern "C" void kernel_launch(void* const* d_in, const int* in_sizes, int n_in,
                              void* d_out, int out_size, void* d_ws, size_t ws_size,
                              hipStream_t stream) {
    const float* x      = (const float*)d_in[0];
    const float* w_qkv  = (const float*)d_in[1];
    const float* w_out  = (const float*)d_in[2];
    const float* w_fc1  = (const float*)d_in[3];
    const float* w_fc2  = (const float*)d_in[4];
    const float* scale1 = (const float*)d_in[5];
    const float* scale2 = (const float*)d_in[6];

    char* ws = (char*)d_ws;
    auto alloc = [&](size_t bytes) -> void* {
        void* p = (void*)ws;
        ws += (bytes + 255) & ~(size_t)255;
        return p;
    };

    const long nQKV = (long)D_MODEL * 3 * D_MODEL;
    const long nOUT = (long)D_MODEL * D_MODEL;
    const long nFC1 = (long)D_MODEL * 4 * D_MODEL;
    const long nFC2 = (long)4 * D_MODEL * D_MODEL;

    bf16_t* wqkv_bf = (bf16_t*)alloc(nQKV * 2);
    bf16_t* wout_bf = (bf16_t*)alloc(nOUT * 2);
    bf16_t* wfc1_bf = (bf16_t*)alloc(nFC1 * 2);
    bf16_t* wfc2_bf = (bf16_t*)alloc(nFC2 * 2);
    bf16_t* xn      = (bf16_t*)alloc((long)M_TOK * D_MODEL * 2);
    bf16_t* qkv_bf  = (bf16_t*)alloc((long)M_TOK * 3 * D_MODEL * 2);
    bf16_t* attn_bf = (bf16_t*)alloc((long)M_TOK * D_MODEL * 2);
    float*  x2      = (float*) alloc((long)M_TOK * D_MODEL * 4);
    bf16_t* xn2     = (bf16_t*)alloc((long)M_TOK * D_MODEL * 2);
    bf16_t* hbuf    = (bf16_t*)alloc((long)M_TOK * 4 * D_MODEL * 2);

    auto cvt = [&](const float* in, bf16_t* out, long n) {
        cvt_f32_bf16<<<dim3((unsigned)((n + 255) / 256)), 256, 0, stream>>>(in, out, n);
    };
    cvt(w_qkv, wqkv_bf, nQKV);
    cvt(w_out, wout_bf, nOUT);
    cvt(w_fc1, wfc1_bf, nFC1);
    cvt(w_fc2, wfc2_bf, nFC2);

    // x -> rmsnorm -> xn (bf16)
    rmsnorm_kernel<<<M_TOK, 256, 0, stream>>>(x, scale1, xn);

    // qkv = xn @ w_qkv  -> bf16 [4096, 3072]
    gemm_bf16_wmma<0><<<dim3(3 * D_MODEL / 128, M_TOK / 128), 256, 0, stream>>>(
        xn, wqkv_bf, M_TOK, 3 * D_MODEL, D_MODEL, nullptr, nullptr, qkv_bf);

    // flash attention -> attn_bf [4096, 1024]
    attention_wmma<<<dim3(SEQ_T / 128, BATCH * N_HEAD), 256, 0, stream>>>(qkv_bf, attn_bf);

    // x2 = x + attn @ w_out  (fp32)
    gemm_bf16_wmma<1><<<dim3(D_MODEL / 128, M_TOK / 128), 256, 0, stream>>>(
        attn_bf, wout_bf, M_TOK, D_MODEL, D_MODEL, x, x2, nullptr);

    // xn2 = rmsnorm(x2, scale2)
    rmsnorm_kernel<<<M_TOK, 256, 0, stream>>>(x2, scale2, xn2);

    // h = gelu(xn2 @ w_fc1) -> bf16 [4096, 4096]
    gemm_bf16_wmma<2><<<dim3(4 * D_MODEL / 128, M_TOK / 128), 256, 0, stream>>>(
        xn2, wfc1_bf, M_TOK, 4 * D_MODEL, D_MODEL, nullptr, nullptr, hbuf);

    // out = x2 + h @ w_fc2  (fp32, final)
    gemm_bf16_wmma<1><<<dim3(D_MODEL / 128, M_TOK / 128), 256, 0, stream>>>(
        hbuf, wfc2_bf, M_TOK, D_MODEL, 4 * D_MODEL, x2, (float*)d_out, nullptr);
}
